// AttnBlock_28114855919740
// MI455X (gfx1250) — matrix operand order repaired and checked by reading, not compiled, not run on hardware
//
#include <hip/hip_runtime.h>
#include <hip/hip_bf16.h>

// ---------------------------------------------------------------------------
// AttnBlock (GroupNorm -> QKV 1x1 conv -> softmax attention -> proj -> +x)
// MI455X / gfx1250, wave32.  WMMA f32_16x16x32_f16 for all GEMMs, flash
// attention (online softmax, scores never leave LDS), Tensor Data Mover for
// all LDS tile staging with double buffering (TENSORcnt-synchronized).
// Sizes: B=4, C=512, N=64*64=4096, GROUPS=32.
// ---------------------------------------------------------------------------

typedef _Float16 v16h __attribute__((ext_vector_type(16)));
typedef _Float16 v8h  __attribute__((ext_vector_type(8)));
typedef _Float16 v4h  __attribute__((ext_vector_type(4)));
typedef float    v8f  __attribute__((ext_vector_type(8)));
typedef float    v4f  __attribute__((ext_vector_type(4)));
typedef int      v4i  __attribute__((ext_vector_type(4)));
typedef unsigned u32x4 __attribute__((ext_vector_type(4)));
typedef int      i32x8 __attribute__((ext_vector_type(8)));
typedef int      i32x4 __attribute__((ext_vector_type(4)));

#define Cc   512
#define Nn   4096
#define Bb   4

#if defined(__has_builtin)
#if __has_builtin(__builtin_amdgcn_tensor_load_to_lds) && \
    __has_builtin(__builtin_amdgcn_s_wait_tensorcnt)
#define USE_TDM 1
#endif
#endif
#ifndef USE_TDM
#define USE_TDM 0
#endif

__device__ inline v8f wmma32(v16h a, v16h b, v8f c) {
  return __builtin_amdgcn_wmma_f32_16x16x32_f16(false, a, false, b, (short)0, c,
                                                false, false);
}

__device__ inline v8f zero8() {
  v8f z = {0.f, 0.f, 0.f, 0.f, 0.f, 0.f, 0.f, 0.f};
  return z;
}

// ---------------------------------------------------------------------------
// Tensor Data Mover: 2D f16 tile (rows x rowlen halves, row stride in halves)
// from global to LDS at byte offset lds_off.  LDS pitch padding via D# pad
// fields: pad_int_code -> 2^(code+1) DWORDs between pads, pad_amt_code ->
// (code+1) DWORDs inserted.  D# layout per CDNA5 ISA 08_async_tensor §8.3/8.4.
// clang-23 toolchain: 6-arg builtin (g0 v4u, g1 v8i, g2 v4i, g3 v4i, v8i, cpol)
// ---------------------------------------------------------------------------
__device__ inline void tdm_load_2d(unsigned lds_off, const _Float16* gsrc,
                                   unsigned rowlen, unsigned rows,
                                   unsigned stride_halves,
                                   unsigned pad_int_code, unsigned pad_amt_code) {
#if USE_TDM
  unsigned long long ga = (unsigned long long)(const void*)gsrc;
  u32x4 g0;
  g0[0] = 1u;                                         // count=1, user D#
  g0[1] = lds_off;                                    // lds_addr (bytes)
  g0[2] = (unsigned)ga;                               // global_addr[31:0]
  g0[3] = (unsigned)((ga >> 32) & 0x1FFFFFFu) | (2u << 30);  // [56:32]|type=2
  i32x8 g1;
  g1[0] = (int)((1u << 16) |                          // data_size = 2 bytes
                (1u << 20) |                          // pad_enable
                (pad_int_code << 22) | (pad_amt_code << 25));
  g1[1] = (int)((rowlen & 0xFFFFu) << 16);            // tensor_dim0 lo
  g1[2] = (int)((rowlen >> 16) | ((rows & 0xFFFFu) << 16));  // d0 hi | d1 lo
  g1[3] = (int)((rows >> 16) | (rowlen << 16));       // d1 hi | tile_dim0
  g1[4] = (int)rows;                                  // tile_dim1 (tile_dim2=0)
  g1[5] = (int)stride_halves;                         // tensor_dim0_stride lo
  g1[6] = 0;                                          // stride hi | d1stride lo
  g1[7] = 0;
  i32x4 gz = {0, 0, 0, 0};
  i32x8 gz8 = {0, 0, 0, 0, 0, 0, 0, 0};
  __builtin_amdgcn_tensor_load_to_lds(g0, g1, gz, gz, gz8, 0);
#else
  (void)lds_off; (void)gsrc; (void)rowlen; (void)rows;
  (void)stride_halves; (void)pad_int_code; (void)pad_amt_code;
#endif
}

// A-fragment (16x32, M=rows): lane L<16 holds row M=L, elems 0..7 -> K 0..7,
// elems 8..15 -> K 16..23; lanes 16..31 same rows, K offset +8.
__device__ inline v16h load_a(const _Float16* t, int row0, int k0, int pitch) {
  int l = threadIdx.x & 31;
  const _Float16* p = t + (row0 + (l & 15)) * pitch + k0 + ((l >> 4) << 3);
  v8h lo = *(const v8h*)p;
  v8h hi = *(const v8h*)(p + 16);
  v16h r;
#pragma unroll
  for (int e = 0; e < 8; ++e) { r[e] = lo[e]; r[e + 8] = hi[e]; }
  return r;
}

// B-fragment (32x16, N=cols): lane L<16 holds col N=L, elems 0..7 -> K 0..7,
// elems 8..15 -> K 16..23; lanes 16..31 same cols, K offset +8.
__device__ inline v16h load_b(const _Float16* t, int col0, int k0, int pitch) {
  int l = threadIdx.x & 31;
  const _Float16* p = t + (col0 + (l & 15)) * pitch + k0 + ((l >> 4) << 3);
  v8h lo = *(const v8h*)p;
  v8h hi = *(const v8h*)(p + 16);
  v16h r;
#pragma unroll
  for (int e = 0; e < 8; ++e) { r[e] = lo[e]; r[e + 8] = hi[e]; }
  return r;
}

// ---------------------------------------------------------------------------
// Kernel 0: convert the four 512x512 f32 weights to f16 (TDM-friendly).
// Wh layout: [z][o][c], z in {q,k,v,p}.
// ---------------------------------------------------------------------------
__global__ void __launch_bounds__(256) wcvt_kernel(
    const float* __restrict__ Wq, const float* __restrict__ Wk,
    const float* __restrict__ Wv, const float* __restrict__ Wp,
    _Float16* __restrict__ Wh) {
  int idx = blockIdx.x * 256 + threadIdx.x;  // 0..262143 (v4 chunks)
  int z = idx >> 16;
  int off = (idx & 65535) * 4;
  const float* s = (z == 0) ? Wq : (z == 1) ? Wk : (z == 2) ? Wv : Wp;
  v4f f = *(const v4f*)(s + off);
  v4h o;
#pragma unroll
  for (int j = 0; j < 4; ++j) o[j] = (_Float16)f[j];
  *(v4h*)(Wh + (size_t)z * (Cc * Cc) + off) = o;
}

// ---------------------------------------------------------------------------
// Kernel 1: GroupNorm statistics. One block per (batch, group).
// ---------------------------------------------------------------------------
__global__ void __launch_bounds__(256) gn_stats_kernel(
    const float* __restrict__ x, float* __restrict__ mu, float* __restrict__ rs) {
  const int bg = blockIdx.x;            // b*32 + g
  const int b = bg >> 5, g = bg & 31;
  const float* p = x + ((size_t)b * Cc + g * 16) * Nn;
  float s = 0.f, s2 = 0.f;
  for (int i = threadIdx.x; i < 16 * Nn; i += 256) {
    float v = p[i];
    s += v; s2 += v * v;
  }
  __shared__ float r1[256], r2[256];
  const int t = threadIdx.x;
  r1[t] = s; r2[t] = s2;
  __syncthreads();
  for (int off = 128; off > 0; off >>= 1) {
    if (t < off) { r1[t] += r1[t + off]; r2[t] += r2[t + off]; }
    __syncthreads();
  }
  if (t == 0) {
    float m = r1[0] * (1.f / 65536.f);
    float v = r2[0] * (1.f / 65536.f) - m * m;
    mu[bg] = m;
    rs[bg] = rsqrtf(v + 1e-6f);
  }
}

// ---------------------------------------------------------------------------
// Kernel 2: apply GroupNorm and transpose to n-major f16: hT[b][n][c].
// ---------------------------------------------------------------------------
__global__ void __launch_bounds__(256) gn_apply_kernel(
    const float* __restrict__ x, const float* __restrict__ gamma,
    const float* __restrict__ beta, const float* __restrict__ mu,
    const float* __restrict__ rs, _Float16* __restrict__ hT) {
  const int n0 = blockIdx.x * 64, c0 = blockIdx.y * 64, b = blockIdx.z;
  __shared__ _Float16 T[64 * 72];
  const int t = threadIdx.x;
#pragma unroll
  for (int i = 0; i < 16; ++i) {
    int e = t + i * 256;
    int cl = e >> 6, nl = e & 63;
    int c = c0 + cl;
    float v = x[((size_t)b * Cc + c) * Nn + n0 + nl];
    int bg = b * 32 + (c >> 4);
    float xn = (v - mu[bg]) * rs[bg] * gamma[c] + beta[c];
    T[cl * 72 + nl] = (_Float16)xn;
  }
  __syncthreads();
#pragma unroll
  for (int i = 0; i < 2; ++i) {
    int e = t + i * 256;
    int nrow = e >> 3, cseg = e & 7;
    v8h pk;
#pragma unroll
    for (int j = 0; j < 8; ++j) pk[j] = T[(cseg * 8 + j) * 72 + nrow];
    *(v8h*)(hT + ((size_t)b * Nn + n0 + nrow) * Cc + c0 + cseg * 8) = pk;
  }
}

// ---------------------------------------------------------------------------
// Kernel 3: QKV projections (M = out-channel o, N = spatial n).
// TDM double-buffered staging of W (64x32) and hT (128x32) chunk tiles.
// Dynamic LDS: 2*(64*40 + 128*40) halves = 30720 bytes.
// ---------------------------------------------------------------------------
#define WLSZ (64 * 40)    // halves per W buffer
#define HLSZ (128 * 40)   // halves per H buffer
#define QKV_SMEM ((2 * WLSZ + 2 * HLSZ) * 2)

__global__ void __launch_bounds__(256) qkv_kernel(
    const _Float16* __restrict__ hT, const _Float16* __restrict__ Wh,
    const float* __restrict__ bq, const float* __restrict__ bk,
    const float* __restrict__ bv,
    _Float16* __restrict__ qT, _Float16* __restrict__ kT,
    _Float16* __restrict__ vB) {
  extern __shared__ __align__(16) char smem[];
  _Float16* Wl = (_Float16*)smem;            // 2 x 64x40
  _Float16* Hl = Wl + 2 * WLSZ;              // 2 x 128x40
  const int t = threadIdx.x, l = t & 31, w = t >> 5;
  const int nb = blockIdx.x * 128, ob = blockIdx.y * 64, z = blockIdx.z;
  const _Float16* Wz = Wh + (size_t)z * (Cc * Cc);
  const float* bias = (z == 0) ? bq : (z == 1) ? bk : bv;

  const int wo = w & 1, wn = w >> 1;  // wave tile: 32o x 32n
  v8f acc[2][2];
#pragma unroll
  for (int i = 0; i < 2; ++i)
#pragma unroll
    for (int j = 0; j < 2; ++j) acc[i][j] = zero8();

#if USE_TDM
  if (w == 0) {  // prologue: chunk 0 into buffer 0
    tdm_load_2d(0, Wz + (size_t)ob * Cc, 32, 64, Cc, 3, 3);
    tdm_load_2d(2 * WLSZ * 2, hT + (size_t)nb * Cc, 32, 128, Cc, 3, 3);
  }
#endif

  for (int kc = 0; kc < 16; ++kc) {
    const int cur = kc & 1;
#if USE_TDM
    if (w == 0) {
      if (kc + 1 < 16) {
        const int nx = (kc + 1) & 1;
        tdm_load_2d((unsigned)(nx * WLSZ * 2),
                    Wz + (size_t)ob * Cc + (kc + 1) * 32, 32, 64, Cc, 3, 3);
        tdm_load_2d((unsigned)((2 * WLSZ + nx * HLSZ) * 2),
                    hT + (size_t)nb * Cc + (kc + 1) * 32, 32, 128, Cc, 3, 3);
        __builtin_amdgcn_s_wait_tensorcnt(2);
      } else {
        __builtin_amdgcn_s_wait_tensorcnt(0);
      }
    }
#else
    {
      int row = t >> 2, seg = t & 3;
      *(v4i*)(&Wl[cur * WLSZ + row * 40 + seg * 8]) =
          *(const v4i*)(Wz + (size_t)(ob + row) * Cc + kc * 32 + seg * 8);
    }
#pragma unroll
    for (int i = 0; i < 2; ++i) {
      int e = t + i * 256;
      int row = e >> 2, seg = e & 3;
      *(v4i*)(&Hl[cur * HLSZ + row * 40 + seg * 8]) =
          *(const v4i*)(hT + (size_t)(nb + row) * Cc + kc * 32 + seg * 8);
    }
#endif
    __syncthreads();
    const _Float16* Wt = Wl + cur * WLSZ;
    const _Float16* Ht = Hl + cur * HLSZ;
    v16h a[2], bf[2];
#pragma unroll
    for (int oi = 0; oi < 2; ++oi) a[oi] = load_a(Wt, wo * 32 + oi * 16, 0, 40);
#pragma unroll
    for (int ni = 0; ni < 2; ++ni) bf[ni] = load_b(Ht, wn * 32 + ni * 16, 0, 40);
#pragma unroll
    for (int oi = 0; oi < 2; ++oi)
#pragma unroll
      for (int ni = 0; ni < 2; ++ni)
        acc[oi][ni] = wmma32(a[oi], bf[ni], acc[oi][ni]);
    __syncthreads();
  }

#pragma unroll
  for (int oi = 0; oi < 2; ++oi) {
#pragma unroll
    for (int ni = 0; ni < 2; ++ni) {
      int o0 = ob + wo * 32 + oi * 16 + ((l >> 4) << 3);  // 8 consecutive o
      int ng = nb + wn * 32 + ni * 16 + (l & 15);         // flat b*N+n
      v4f b0 = *(const v4f*)(bias + o0);
      v4f b1 = *(const v4f*)(bias + o0 + 4);
      if (z < 2) {
        v8h r;
#pragma unroll
        for (int e = 0; e < 4; ++e) {
          r[e]     = (_Float16)(acc[oi][ni][e]     + b0[e]);
          r[e + 4] = (_Float16)(acc[oi][ni][e + 4] + b1[e]);
        }
        _Float16* dst = (z == 0) ? qT : kT;
        *(v8h*)(dst + (size_t)ng * Cc + o0) = r;
      } else {
        int bi = ng >> 12, nl2 = ng & 4095;
        _Float16* dst = vB + ((size_t)bi * Cc + o0) * Nn + nl2;
#pragma unroll
        for (int e = 0; e < 4; ++e) {
          dst[(size_t)e * Nn]       = (_Float16)(acc[oi][ni][e]     + b0[e]);
          dst[(size_t)(e + 4) * Nn] = (_Float16)(acc[oi][ni][e + 4] + b1[e]);
        }
      }
    }
  }
}

// ---------------------------------------------------------------------------
// Kernel 4: flash attention.  One block = (batch, 64-query tile), 8 waves.
// TDM streams 32-key K/V tiles (double buffered); S=Q^T K in LDS, online
// softmax, O=V P^T in VGPR accumulators (wave owns 64c x 64n).
// Dynamic LDS 229,376 bytes (< 320 KB/WGP).
// ---------------------------------------------------------------------------
#define QP 520   // Q/K tile pitch in halves (512 + 4-DWORD TDM pad)
#define VP 40    // V/P tile pitch in halves (32 + 4-DWORD TDM pad)
#define SP 33    // score tile pitch in floats
#define KSZ (32 * QP)    // halves per K buffer
#define VSZ (512 * VP)   // halves per V buffer
#define OFF_K0 (64 * QP * 2)
#define OFF_V0 ((64 * QP + 2 * KSZ) * 2)
#define FLASH_SMEM ((64 * QP + 2 * KSZ + 2 * VSZ + 64 * VP) * 2 + \
                    (64 * SP + 3 * 64) * 4)

__global__ void __launch_bounds__(256) flash_kernel(
    const _Float16* __restrict__ qT, const _Float16* __restrict__ kT,
    const _Float16* __restrict__ vB, _Float16* __restrict__ aO) {
  extern __shared__ __align__(16) char smem[];
  _Float16* Qs  = (_Float16*)smem;          // 64 x QP
  _Float16* Ks0 = Qs + 64 * QP;             // 2 x (32 x QP)
  _Float16* Vs0 = Ks0 + 2 * KSZ;            // 2 x (512 x VP)
  _Float16* Ps  = Vs0 + 2 * VSZ;            // 64 x VP
  float*    Ss  = (float*)(Ps + 64 * VP);   // 64 x SP
  float*    Mr  = Ss + 64 * SP;
  float*    Lr  = Mr + 64;
  float*    Rr  = Lr + 64;

  const int t = threadIdx.x, l = t & 31, w = t >> 5;
  const int n0 = blockIdx.x * 64, b = blockIdx.y;
  const size_t qbase = ((size_t)b * Nn + n0) * Cc;

#if USE_TDM
  if (w == 0) {  // prologue: Q tile + K/V tile 0
    tdm_load_2d(0, qT + qbase, Cc, 64, Cc, 7, 3);
    tdm_load_2d(OFF_K0, kT + (size_t)b * Nn * Cc, Cc, 32, Cc, 7, 3);
    tdm_load_2d(OFF_V0, vB + (size_t)b * Cc * Nn, 32, 512, Nn, 3, 3);
  }
#else
#pragma unroll
  for (int i = 0; i < 16; ++i) {
    int e = t + i * 256;
    int row = e >> 6, seg = e & 63;
    *(v4i*)(&Qs[row * QP + seg * 8]) =
        *(const v4i*)(qT + qbase + (size_t)row * Cc + seg * 8);
  }
#endif
  if (t < 64) { Mr[t] = -3.0e38f; Lr[t] = 0.f; }

  v8f acc[4][4];
#pragma unroll
  for (int i = 0; i < 4; ++i)
#pragma unroll
    for (int j = 0; j < 4; ++j) acc[i][j] = zero8();

  const int sr = w & 3, sc = w >> 2;  // this wave's 16x16 S sub-tile

  for (int it = 0; it < Nn / 32; ++it) {
    const int m0 = it * 32;
    const int cur = it & 1;
#if USE_TDM
    if (w == 0) {
      if (it + 1 < Nn / 32) {
        const int nx = (it + 1) & 1;
        tdm_load_2d((unsigned)(OFF_K0 + nx * KSZ * 2),
                    kT + ((size_t)b * Nn + m0 + 32) * Cc, Cc, 32, Cc, 7, 3);
        tdm_load_2d((unsigned)(OFF_V0 + nx * VSZ * 2),
                    vB + (size_t)b * Cc * Nn + m0 + 32, 32, 512, Nn, 3, 3);
        __builtin_amdgcn_s_wait_tensorcnt(2);  // tile `it` (and Q) complete
      } else {
        __builtin_amdgcn_s_wait_tensorcnt(0);
      }
    }
#else
#pragma unroll
    for (int i = 0; i < 8; ++i) {
      int e = t + i * 256;
      int row = e >> 6, seg = e & 63;
      *(v4i*)(&Ks0[cur * KSZ + row * QP + seg * 8]) =
          *(const v4i*)(kT + ((size_t)b * Nn + m0 + row) * Cc + seg * 8);
    }
#pragma unroll
    for (int i = 0; i < 8; ++i) {
      int e = t + i * 256;
      int row = e >> 2, seg = e & 3;
      *(v4i*)(&Vs0[cur * VSZ + row * VP + seg * 8]) =
          *(const v4i*)(vB + ((size_t)b * Cc + row) * Nn + m0 + seg * 8);
    }
#endif
    __syncthreads();
    const _Float16* Kc = Ks0 + cur * KSZ;
    const _Float16* Vc = Vs0 + cur * VSZ;

    // S = (1/sqrt(C)) Q^T K : each wave one 16n x 16m sub-tile over K=512
    v8f s = zero8();
#pragma unroll
    for (int kc = 0; kc < 16; ++kc) {
      v16h aq = load_a(Qs, sr * 16, kc * 32, QP);
      v16h bk2 = load_b(Kc, sc * 16, kc * 32, QP);
      s = wmma32(aq, bk2, s);
    }
#pragma unroll
    for (int r = 0; r < 8; ++r)
      Ss[(sr * 16 + ((l >> 4) << 3) + r) * SP + sc * 16 + (l & 15)] =
          s[r] * 0.04419417382415922f;  // 1/sqrt(512)
    __syncthreads();

    // Online softmax over the 32 new keys (one thread per query row)
    if (t < 64) {
      float mold = Mr[t];
      float mx = mold;
      for (int j = 0; j < 32; ++j) mx = fmaxf(mx, Ss[t * SP + j]);
      float rsc = __expf(mold - mx);
      float sum = 0.f;
      for (int j = 0; j < 32; ++j) {
        float p = __expf(Ss[t * SP + j] - mx);
        sum += p;
        Ps[t * VP + j] = (_Float16)p;
      }
      Lr[t] = Lr[t] * rsc + sum;
      Mr[t] = mx;
      Rr[t] = rsc;
    }
    __syncthreads();

    // Rescale accumulators, then O += V * P^T (wave owns c in [w*64, w*64+64))
    float rscl[4];
#pragma unroll
    for (int ni = 0; ni < 4; ++ni) rscl[ni] = Rr[ni * 16 + (l & 15)];
#pragma unroll
    for (int ci = 0; ci < 4; ++ci)
#pragma unroll
      for (int ni = 0; ni < 4; ++ni)
#pragma unroll
        for (int e = 0; e < 8; ++e) acc[ci][ni][e] *= rscl[ni];

    v16h av[4], bpf[4];
#pragma unroll
    for (int ci = 0; ci < 4; ++ci) av[ci] = load_a(Vc, w * 64 + ci * 16, 0, VP);
#pragma unroll
    for (int ni = 0; ni < 4; ++ni) bpf[ni] = load_b(Ps, ni * 16, 0, VP);
#pragma unroll
    for (int ci = 0; ci < 4; ++ci)
#pragma unroll
      for (int ni = 0; ni < 4; ++ni)
        acc[ci][ni] = wmma32(av[ci], bpf[ni], acc[ci][ni]);
    __syncthreads();  // all waves done with buffers before TDM overwrites
  }

  // Finalize: divide by L, store attnout n-major f16 [b][n][c]
#pragma unroll
  for (int ni = 0; ni < 4; ++ni) {
    float inv = 1.f / Lr[ni * 16 + (l & 15)];
    int n = n0 + ni * 16 + (l & 15);
#pragma unroll
    for (int ci = 0; ci < 4; ++ci) {
      v8h o;
#pragma unroll
      for (int e = 0; e < 8; ++e) o[e] = (_Float16)(acc[ci][ni][e] * inv);
      int c = w * 64 + ci * 16 + ((l >> 4) << 3);
      *(v8h*)(aO + ((size_t)b * Nn + n) * Cc + c) = o;
    }
  }
}

// ---------------------------------------------------------------------------
// Kernel 5: output projection + bias + residual (M = n, N = o; contiguous
// 32B fp32 stores + residual loads per lane).  TDM double-buffered staging.
// ---------------------------------------------------------------------------
__global__ void __launch_bounds__(256) proj_kernel(
    const _Float16* __restrict__ aO, const _Float16* __restrict__ Wh,
    const float* __restrict__ bp, const float* __restrict__ x,
    float* __restrict__ out) {
  extern __shared__ __align__(16) char smem[];
  _Float16* Bl = (_Float16*)smem;            // 2 x 64x40  (Wp rows o)
  _Float16* Al = Bl + 2 * WLSZ;              // 2 x 128x40 (attnout rows n)
  const int t = threadIdx.x, l = t & 31, w = t >> 5;
  const int nb = blockIdx.x * 128, ob = blockIdx.y * 64;
  const int wn = w >> 1, wo = w & 1;
  const _Float16* Wp16 = Wh + (size_t)3 * (Cc * Cc);
  v8f acc[2][2];
#pragma unroll
  for (int i = 0; i < 2; ++i)
#pragma unroll
    for (int j = 0; j < 2; ++j) acc[i][j] = zero8();

#if USE_TDM
  if (w == 0) {
    tdm_load_2d(0, Wp16 + (size_t)ob * Cc, 32, 64, Cc, 3, 3);
    tdm_load_2d(2 * WLSZ * 2, aO + (size_t)nb * Cc, 32, 128, Cc, 3, 3);
  }
#endif

  for (int kc = 0; kc < 16; ++kc) {
    const int cur = kc & 1;
#if USE_TDM
    if (w == 0) {
      if (kc + 1 < 16) {
        const int nx = (kc + 1) & 1;
        tdm_load_2d((unsigned)(nx * WLSZ * 2),
                    Wp16 + (size_t)ob * Cc + (kc + 1) * 32, 32, 64, Cc, 3, 3);
        tdm_load_2d((unsigned)((2 * WLSZ + nx * HLSZ) * 2),
                    aO + (size_t)nb * Cc + (kc + 1) * 32, 32, 128, Cc, 3, 3);
        __builtin_amdgcn_s_wait_tensorcnt(2);
      } else {
        __builtin_amdgcn_s_wait_tensorcnt(0);
      }
    }
#else
    {
      int row = t >> 2, seg = t & 3;
      *(v4i*)(&Bl[cur * WLSZ + row * 40 + seg * 8]) =
          *(const v4i*)(Wp16 + (size_t)(ob + row) * Cc + kc * 32 + seg * 8);
    }
#pragma unroll
    for (int i = 0; i < 2; ++i) {
      int e = t + i * 256;
      int row = e >> 2, seg = e & 3;
      *(v4i*)(&Al[cur * HLSZ + row * 40 + seg * 8]) =
          *(const v4i*)(aO + (size_t)(nb + row) * Cc + kc * 32 + seg * 8);
    }
#endif
    __syncthreads();
    const _Float16* Bt = Bl + cur * WLSZ;
    const _Float16* At = Al + cur * HLSZ;
    v16h a[2], bf[2];
#pragma unroll
    for (int ni = 0; ni < 2; ++ni) a[ni] = load_a(At, wn * 32 + ni * 16, 0, 40);
#pragma unroll
    for (int oi = 0; oi < 2; ++oi) bf[oi] = load_b(Bt, wo * 32 + oi * 16, 0, 40);
#pragma unroll
    for (int ni = 0; ni < 2; ++ni)
#pragma unroll
      for (int oi = 0; oi < 2; ++oi)
        acc[ni][oi] = wmma32(a[ni], bf[oi], acc[ni][oi]);
    __syncthreads();
  }

#pragma unroll
  for (int ni = 0; ni < 2; ++ni) {
#pragma unroll
    for (int oi = 0; oi < 2; ++oi) {
      int nflat = nb + wn * 32 + ni * 16 + ((l >> 4) << 3);  // 8 consecutive n
      int og = ob + wo * 32 + oi * 16 + (l & 15);
      int bi = nflat >> 12, nl2 = nflat & 4095;
      size_t base = ((size_t)bi * Cc + og) * Nn + nl2;
      float bb = bp[og];
      v4f x0 = *(const v4f*)(x + base);
      v4f x1 = *(const v4f*)(x + base + 4);
      v4f r0, r1;
#pragma unroll
      for (int j = 0; j < 4; ++j) {
        r0[j] = x0[j] + acc[ni][oi][j] + bb;
        r1[j] = x1[j] + acc[ni][oi][j + 4] + bb;
      }
      *(v4f*)(out + base) = r0;
      *(v4f*)(out + base + 4) = r1;
    }
  }
}

// ---------------------------------------------------------------------------
// Host launcher
// ---------------------------------------------------------------------------
extern "C" void kernel_launch(void* const* d_in, const int* in_sizes, int n_in,
                              void* d_out, int out_size, void* d_ws, size_t ws_size,
                              hipStream_t stream) {
  (void)in_sizes; (void)n_in; (void)out_size; (void)ws_size;
  const float* x     = (const float*)d_in[0];
  const float* gamma = (const float*)d_in[1];
  const float* beta  = (const float*)d_in[2];
  const float* Wq    = (const float*)d_in[3];
  const float* bq    = (const float*)d_in[4];
  const float* Wk    = (const float*)d_in[5];
  const float* bk    = (const float*)d_in[6];
  const float* Wv    = (const float*)d_in[7];
  const float* bv    = (const float*)d_in[8];
  const float* Wp    = (const float*)d_in[9];
  const float* bp    = (const float*)d_in[10];
  float* out = (float*)d_out;

  char* ws = (char*)d_ws;
  const size_t MAT = (size_t)Bb * Nn * Cc * sizeof(_Float16);  // 16 MiB
  _Float16* hT = (_Float16*)(ws);
  _Float16* qT = (_Float16*)(ws + 1 * MAT);
  _Float16* kT = (_Float16*)(ws + 2 * MAT);
  _Float16* vB = (_Float16*)(ws + 3 * MAT);
  _Float16* aO = (_Float16*)(ws + 4 * MAT);
  _Float16* Wh = (_Float16*)(ws + 5 * MAT);            // 4 x 512 x 512 f16
  float* mu = (float*)(ws + 5 * MAT + 4 * (size_t)Cc * Cc * 2);
  float* rs = mu + Bb * 32;

  wcvt_kernel<<<dim3(1024), 256, 0, stream>>>(Wq, Wk, Wv, Wp, Wh);
  gn_stats_kernel<<<dim3(Bb * 32), 256, 0, stream>>>(x, mu, rs);
  gn_apply_kernel<<<dim3(Nn / 64, Cc / 64, Bb), 256, 0, stream>>>(
      x, gamma, beta, mu, rs, hT);

  (void)hipFuncSetAttribute((const void*)qkv_kernel,
                            hipFuncAttributeMaxDynamicSharedMemorySize,
                            (int)QKV_SMEM);
  qkv_kernel<<<dim3((Bb * Nn) / 128, Cc / 64, 3), 256, QKV_SMEM, stream>>>(
      hT, Wh, bq, bk, bv, qT, kT, vB);

  (void)hipFuncSetAttribute((const void*)flash_kernel,
                            hipFuncAttributeMaxDynamicSharedMemorySize,
                            (int)FLASH_SMEM);
  flash_kernel<<<dim3(Nn / 64, Bb), 256, FLASH_SMEM, stream>>>(qT, kT, vB, aO);

  (void)hipFuncSetAttribute((const void*)proj_kernel,
                            hipFuncAttributeMaxDynamicSharedMemorySize,
                            (int)QKV_SMEM);
  proj_kernel<<<dim3((Bb * Nn) / 128, Cc / 64), 256, QKV_SMEM, stream>>>(
      aO, Wh, bp, x, out);
}
